// Disk_18253611008110
// MI455X (gfx1250) — matrix operand order repaired and checked by reading, not compiled, not run on hardware
//
#include <hip/hip_runtime.h>
#include <hip/hip_bf16.h>

// ---------------------------------------------------------------------------
// DISK/SuperPoint mutual-NN matcher for MI455X (gfx1250, wave32, WMMA).
//
// dist = sqrt2*sqrt(clip(1 - S, 1e-6)) is monotone-decreasing in S = fs0@fs1^T,
// so NN / ratio tests are done on similarities; the 256MB dist matrix is never
// materialized. Fused f16-WMMA GEMM + top-2 reduction, run twice (fwd / bwd).
// ---------------------------------------------------------------------------

typedef _Float16 h8  __attribute__((ext_vector_type(8)));
typedef _Float16 h16 __attribute__((ext_vector_type(16)));
typedef float    f8  __attribute__((ext_vector_type(8)));

constexpr int F_DIM   = 256;  // descriptor dim (K)
constexpr int BM      = 128;  // rows per workgroup
constexpr int BN_TILE = 128;  // cols per inner tile
constexpr int NSTRIPS = 8;    // column strips (grid.x)
constexpr int KSTRIDE = 264;  // LDS halves per row: mult of 8 -> 16B-aligned b128 reads

struct Top2 { float v1; int i1; float v2; int i2; };

__device__ __forceinline__ void top2_merge(float v, int i,
                                           float& V1, int& I1,
                                           float& V2, int& I2) {
  if (v > V1 || (v == V1 && (unsigned)i < (unsigned)I1)) {
    V2 = V1; I2 = I1; V1 = v; I1 = i;
  } else if (v > V2) {          // v == V1 duplicate falls here -> recorded as 2nd
    V2 = v; I2 = i;
  }
}

// A: [F][NA] row-major (descriptors are stored feature-major: d[f*NA + n])
// B: [F][NB] row-major
// Computes, for every row n in this block's 128-row strip, the top-2 of
// S[n, m] = sum_f A[f,n]*B[f,m] over this block's 1024-column strip.
__global__ __launch_bounds__(256) void match_gemm_top2(
    const float* __restrict__ A, const float* __restrict__ B,
    int NA, int NB, int cols_per_strip,
    Top2* __restrict__ partial /* [NSTRIPS][NA] */) {
  extern __shared__ _Float16 smem[];
  _Float16* ldsA = smem;                 // [BM][KSTRIDE]  (row-major, k contiguous)
  _Float16* ldsB = smem + BM * KSTRIDE;  // [BN_TILE][KSTRIDE] (B^T, k contiguous)

  const int tid   = threadIdx.x;
  const int wave  = tid >> 5;
  const int lane  = tid & 31;
  const int lrow  = lane & 15;  // M (A/D) or N (B) index within a 16x16 tile
  const int lhalf = lane >> 4;  // K-half selector per WMMA fragment layouts

  const int n0     = blockIdx.y * BM;
  const int strip  = blockIdx.x;
  const int m_base = strip * cols_per_strip;
  const int ntiles = cols_per_strip / BN_TILE;

  // Stage A strip [128 rows x 256 K] -> LDS f16, transposed to [row][k].
  // Global reads are n-contiguous (coalesced); f32 -> f16 convert in flight.
  for (int i = tid; i < BM * F_DIM; i += 256) {
    const int f = i >> 7;         // BM == 128
    const int r = i & (BM - 1);
    ldsA[r * KSTRIDE + f] = (_Float16)A[(size_t)f * NA + n0 + r];
  }

  float tv1[8], tv2[8]; int ti1[8], ti2[8];
#pragma unroll
  for (int g = 0; g < 8; ++g) { tv1[g] = -3.4e38f; tv2[g] = -3.4e38f; ti1[g] = -1; ti2[g] = -1; }

  for (int t = 0; t < ntiles; ++t) {
    const int m0 = m_base + t * BN_TILE;
    __syncthreads();  // previous tile fully consumed (and A staged for t==0)
    for (int i = tid; i < BN_TILE * F_DIM; i += 256) {
      const int f = i >> 7;
      const int c = i & (BN_TILE - 1);
      ldsB[c * KSTRIDE + f] = (_Float16)B[(size_t)f * NB + m0 + c];
    }
    __syncthreads();

    // Prefetch next B tile toward L2 (global_prefetch_b8); tid == f row.
    if (t + 1 < ntiles) {
      const float* nb = &B[(size_t)tid * NB + m0 + BN_TILE];
      __builtin_prefetch(nb, 0, 0);
      __builtin_prefetch(nb + 32, 0, 0);
      __builtin_prefetch(nb + 64, 0, 0);
      __builtin_prefetch(nb + 96, 0, 0);
    }

    f8 acc[8] = {};
    const int arow = wave * 16 + lrow;
#pragma unroll
    for (int kt = 0; kt < F_DIM / 32; ++kt) {
      const int kb = kt * 32;
      // A fragment 16x32 f16 (ISA 7.12.2): lane {M=lrow, khalf=lhalf};
      // VGPR0-3 <- K = khalf*8 + 0..7, VGPR4-7 <- K = 16 + khalf*8 + 0..7.
      const _Float16* ap = &ldsA[arow * KSTRIDE + kb + lhalf * 8];
      const h8 alo = *(const h8*)ap;
      const h8 ahi = *(const h8*)(ap + 16);
      const h16 af = __builtin_shufflevector(alo, ahi,
          0, 1, 2, 3, 4, 5, 6, 7, 8, 9, 10, 11, 12, 13, 14, 15);
#pragma unroll
      for (int c = 0; c < 8; ++c) {
        // B fragment 32x16 f16 (extrapolated from ISA sparse 64x16 layout):
        // lane {N=lrow, khalf=lhalf}; VGPR0-3 <- K = khalf*8 + 0..7, VGPR4-7 <- K = 16 + khalf*8 + 0..7.
        const _Float16* bp = &ldsB[(c * 16 + lrow) * KSTRIDE + kb + lhalf * 8];
        const h8 blo = *(const h8*)bp;
        const h8 bhi = *(const h8*)(bp + 16);
        const h16 bf = __builtin_shufflevector(blo, bhi,
            0, 1, 2, 3, 4, 5, 6, 7, 8, 9, 10, 11, 12, 13, 14, 15);
        acc[c] = __builtin_amdgcn_wmma_f32_16x16x32_f16(
            false, af, false, bf, (short)0, acc[c], false, false);
      }
    }

    // D layout: lane holds column N=lrow; VGPR g -> row M = lhalf*8 + g.
    // Each lane updates its private top-2 for the 8 rows it owns.
#pragma unroll
    for (int c = 0; c < 8; ++c) {
      const int colg = m0 + c * 16 + lrow;
#pragma unroll
      for (int g = 0; g < 8; ++g)
        top2_merge(acc[c][g], colg, tv1[g], ti1[g], tv2[g], ti2[g]);
    }
  }

  // Cross-lane merge: 16 lanes hold candidates per row. Funnel through LDS.
  __syncthreads();
  float* smf = (float*)smem;
  int*   smi = (int*)smem;
#pragma unroll
  for (int g = 0; g < 8; ++g) {
    const int row  = wave * 16 + lhalf * 8 + g;
    const int slot = (row * 16 + lrow) * 4;
    smf[slot]     = tv1[g];
    smi[slot + 1] = ti1[g];
    smf[slot + 2] = tv2[g];
    smi[slot + 3] = ti2[g];
  }
  __syncthreads();
  if (tid < BM) {
    float V1 = -3.4e38f, V2 = -3.4e38f; int I1 = -1, I2 = -1;
    for (int l = 0; l < 16; ++l) {
      const int slot = (tid * 16 + l) * 4;
      top2_merge(smf[slot],     smi[slot + 1], V1, I1, V2, I2);
      top2_merge(smf[slot + 2], smi[slot + 3], V1, I1, V2, I2);
    }
    Top2 o; o.v1 = V1; o.i1 = I1; o.v2 = V2; o.i2 = I2;
    partial[(size_t)strip * NA + n0 + tid] = o;
  }
}

__global__ __launch_bounds__(256) void combine_top2(
    const Top2* __restrict__ part, int nrows,
    int* __restrict__ nn, float* __restrict__ s1, float* __restrict__ s2) {
  const int r = blockIdx.x * 256 + threadIdx.x;
  if (r >= nrows) return;
  float V1 = -3.4e38f, V2 = -3.4e38f; int I1 = -1, I2 = -1;
  for (int s = 0; s < NSTRIPS; ++s) {
    const Top2 p = part[(size_t)s * nrows + r];
    top2_merge(p.v1, p.i1, V1, I1, V2, I2);
    top2_merge(p.v2, p.i2, V1, I1, V2, I2);
  }
  nn[r] = I1; s1[r] = V1; s2[r] = V2;
}

__global__ __launch_bounds__(256) void finalize_matches(
    const int* __restrict__ fnn, const float* __restrict__ fs1, const float* __restrict__ fs2,
    const int* __restrict__ bnn, const float* __restrict__ bs1, const float* __restrict__ bs2,
    int N, int M,
    int* __restrict__ indices0, int* __restrict__ matches1,
    int* __restrict__ mscores0, float* __restrict__ mscores1) {
  const int i = blockIdx.x * 256 + threadIdx.x;
  if (i < M) { matches1[i] = -1; mscores1[i] = 0.0f; }
  if (i < N) {
    const int j = fnn[i];
    // ratio test in dist space: sqrt2*sqrt(d) monotone, so compare clipped 1-s.
    const float fd1 = fmaxf(1.0f - fs1[i], 1e-6f);
    const float fd2 = fmaxf(1.0f - fs2[i], 1e-6f);
    bool mutual = false;
    if (fd1 < fd2 && j >= 0) {
      const float bd1 = fmaxf(1.0f - bs1[j], 1e-6f);
      const float bd2 = fmaxf(1.0f - bs2[j], 1e-6f);
      mutual = (bd1 < bd2) && (bnn[j] == i);
    }
    const int idx = mutual ? j : -1;
    indices0[i] = idx;
    mscores0[i] = (idx > 0) ? 1 : 0;  // reference uses strict '> 0'
  }
}

extern "C" void kernel_launch(void* const* d_in, const int* in_sizes, int n_in,
                              void* d_out, int out_size, void* d_ws, size_t ws_size,
                              hipStream_t stream) {
  const float* d0 = (const float*)d_in[0];  // [1, F, N] == [F][N]
  const float* d1 = (const float*)d_in[1];  // [1, F, M] == [F][M]
  const int N = in_sizes[0] / F_DIM;        // 8192
  const int M = in_sizes[1] / F_DIM;        // 8192

  // Workspace carve-up (~2.3 MB).
  char* ws = (char*)d_ws;
  Top2* fwdP = (Top2*)ws;  ws += (size_t)NSTRIPS * N * sizeof(Top2);
  Top2* bckP = (Top2*)ws;  ws += (size_t)NSTRIPS * M * sizeof(Top2);
  int*   fnn = (int*)ws;   ws += (size_t)N * sizeof(int);
  float* fs1 = (float*)ws; ws += (size_t)N * sizeof(float);
  float* fs2 = (float*)ws; ws += (size_t)N * sizeof(float);
  int*   bnn = (int*)ws;   ws += (size_t)M * sizeof(int);
  float* bs1 = (float*)ws; ws += (size_t)M * sizeof(float);
  float* bs2 = (float*)ws; ws += (size_t)M * sizeof(float);

  // Output segments (flat concat, all 4-byte elements).
  int*   indices0 = (int*)d_out;
  int*   matches1 = indices0 + N;
  int*   mscores0 = matches1 + M;
  float* mscores1 = (float*)(mscores0 + N);

  const int shmem = 2 * BM * KSTRIDE * (int)sizeof(_Float16);  // 132 KB < 320 KB/WGP
  (void)hipFuncSetAttribute((const void*)match_gemm_top2,
                            hipFuncAttributeMaxDynamicSharedMemorySize, shmem);

  const int cps_f = M / NSTRIPS;  // 1024
  const int cps_b = N / NSTRIPS;

  // Forward: rows = keypoints0, cols = keypoints1.
  match_gemm_top2<<<dim3(NSTRIPS, N / BM), 256, shmem, stream>>>(
      d0, d1, N, M, cps_f, fwdP);
  // Backward: rows = keypoints1, cols = keypoints0.
  match_gemm_top2<<<dim3(NSTRIPS, M / BM), 256, shmem, stream>>>(
      d1, d0, M, N, cps_b, bckP);

  combine_top2<<<(N + 255) / 256, 256, 0, stream>>>(fwdP, N, fnn, fs1, fs2);
  combine_top2<<<(M + 255) / 256, 256, 0, stream>>>(bckP, M, bnn, bs1, bs2);

  const int NM = (N > M) ? N : M;
  finalize_matches<<<(NM + 255) / 256, 256, 0, stream>>>(
      fnn, fs1, fs2, bnn, bs1, bs2, N, M,
      indices0, matches1, mscores0, mscores1);
}
